// NegMarginalLogLikelihood_deep_34394098106752
// MI455X (gfx1250) — matrix operand order repaired and checked by reading, not compiled, not run on hardware
//
#include <hip/hip_runtime.h>
#include <hip/hip_bf16.h>

typedef __attribute__((ext_vector_type(16))) _Float16 v16h;
typedef __attribute__((ext_vector_type(8)))  _Float16 v8h;
typedef __attribute__((ext_vector_type(2)))  _Float16 h2;
typedef __attribute__((ext_vector_type(8)))  float    v8f;

#define GP_M      256
#define PI_F      3.14159265358979323846f

// workspace layout (float indices)
#define WS_MAX     0   // max(m_train), stored as float bits (all inputs >= 0)
#define WS_YTY     1
#define WS_L       2
#define WS_ALPHA   3   // 0.5*pi/L
#define WS_ISL     4   // 1/sqrt(L)
#define INVLAM_OFF 8
#define PHITY_OFF  (INVLAM_OFF + 256)          // 264
#define Z_OFF      (PHITY_OFF + 256)           // 520
#define WS_TOTAL   (Z_OFF + 256 * 256)         // 66056 floats (~258 KB)

// ---------------------------------------------------------------------------
// 0) zero the workspace (harness poisons it with 0xAA)
// ---------------------------------------------------------------------------
__global__ void gp_init_kernel(float* ws) {
    int i = blockIdx.x * blockDim.x + threadIdx.x;
    if (i < WS_TOTAL) ws[i] = 0.0f;
}

// ---------------------------------------------------------------------------
// 1) max(m_train) and y.y  (grid-stride + LDS tree + global atomics)
// ---------------------------------------------------------------------------
__global__ void gp_reduce_kernel(const float* __restrict__ m_train,
                                 const float* __restrict__ y_train,
                                 float* ws, int n) {
    __shared__ float smax[256];
    __shared__ float ssum[256];
    int t = threadIdx.x;
    int idx = blockIdx.x * blockDim.x + t;
    int stride = gridDim.x * blockDim.x;
    float mx = 0.0f, s = 0.0f;
    for (int i = idx; i < n; i += stride) {
        mx = fmaxf(mx, m_train[i]);
        float yv = y_train[i];
        s += yv * yv;
    }
    smax[t] = mx; ssum[t] = s;
    for (int o = 128; o > 0; o >>= 1) {
        __syncthreads();
        if (t < o) { smax[t] = fmaxf(smax[t], smax[t + o]); ssum[t] += ssum[t + o]; }
    }
    if (t == 0) {
        atomicMax((unsigned int*)&ws[WS_MAX], __float_as_uint(smax[0]));  // valid: all >= 0
        atomicAdd(&ws[WS_YTY], ssum[0]);
    }
}

// ---------------------------------------------------------------------------
// 2) scalars: L, alpha, 1/sqrt(L); spectral weights inv_lambda[256]
// ---------------------------------------------------------------------------
__global__ void gp_prep_kernel(const float* sigma_f, const float* lengthscale,
                               const float* sigma_n, float* ws) {
    __shared__ float sL;
    int t = threadIdx.x;
    float ls = lengthscale[0];
    if (t == 0) {
        float mx = __uint_as_float(((const unsigned int*)ws)[WS_MAX]);
        float L = fmaxf(1.5f * mx, PI_F * (float)GP_M * fabsf(ls) / (2.0f * 3.5f));
        ws[WS_L]     = L;
        ws[WS_ALPHA] = 0.5f * PI_F / L;
        ws[WS_ISL]   = rsqrtf(L);
        sL = L;
    }
    __syncthreads();
    float L  = sL;
    float sf = sigma_f[0];
    float omega = PI_F * (float)(t + 1) / (2.0f * L);
    float invlam = (1.0f / (sf * sf)) * rsqrtf(2.0f * PI_F * ls * ls)
                 * __expf(0.5f * ls * ls * omega * omega);
    ws[INVLAM_OFF + t] = invlam;
    (void)sigma_n;
}

// ---------------------------------------------------------------------------
// 3) Z = phi^T phi via v_wmma_f32_16x16x32_f16.
//    grid = (KSPLIT=32, 4 quadrants). block = 256 threads = 8 wave32s.
//    LDS tile is TRANSPOSED + fragment-major: sphiT[j][n], row stride 40
//    halves (80 B, 16 B aligned) so A and B fragments are 2x ds_load_b128
//    per lane instead of 16 scalar u16 loads.
// ---------------------------------------------------------------------------
__global__ void __launch_bounds__(256)
gp_gemm_kernel(const float* __restrict__ m_train, float* __restrict__ ws, int n) {
    __shared__ _Float16 sphiT[256][40];   // j-major; 40 = 32 n + pad (16B-aligned rows)

    const int tid  = threadIdx.x;
    const int wave = tid >> 5;            // wave32
    const int lane = tid & 31;
    const int half = lane >> 4;
    const int mn   = lane & 15;

    const int sub = blockIdx.y;           // quadrant of Z
    const int rh  = sub >> 1, ch = sub & 1;
    const int rowbase = (rh * 8 + wave) * 16;

    const float L     = ws[WS_L];
    const float alpha = ws[WS_ALPHA];
    const float isl   = ws[WS_ISL];
    float* Zg = ws + Z_OFF;

    v8f acc[8];
#pragma unroll
    for (int t = 0; t < 8; t++)
#pragma unroll
        for (int r = 0; r < 8; r++) acc[t][r] = 0.0f;

    const int nPer = n / gridDim.x;       // 4096 for KSPLIT=32
    const int nBeg = blockIdx.x * nPer;
    const int fill_np = tid >> 4;         // 0..15 : n-pair (2n, 2n+1)
    const int fill_jb = (tid & 15) * 16;  // 16 consecutive j per thread

    for (int n0 = nBeg; n0 < nBeg + nPer; n0 += 32) {
        // ---- generate phi[j][n0..n0+31] in f16 (two sin recurrences/thread,
        //      packed b32 stores: adjacent n are adjacent halves) ------------
        {
            float x0 = m_train[n0 + 2 * fill_np];
            float x1 = m_train[n0 + 2 * fill_np + 1];
            float th0 = alpha * (x0 + L);
            float th1 = alpha * (x1 + L);
            float jb1 = (float)(fill_jb + 1);
            float s0  = __sinf(jb1 * th0), c0  = __cosf(jb1 * th0);
            float sd0 = __sinf(th0),       cd0 = __cosf(th0);
            float s1  = __sinf(jb1 * th1), c1  = __cosf(jb1 * th1);
            float sd1 = __sinf(th1),       cd1 = __cosf(th1);
#pragma unroll
            for (int jj = 0; jj < 16; jj++) {
                h2 pkt;
                pkt[0] = (_Float16)(s0 * isl);
                pkt[1] = (_Float16)(s1 * isl);
                *(h2*)&sphiT[fill_jb + jj][2 * fill_np] = pkt;
                float t0 = s0 * cd0 + c0 * sd0;   // sin((j+1)*theta)
                c0 = c0 * cd0 - s0 * sd0;
                s0 = t0;
                float t1 = s1 * cd1 + c1 * sd1;
                c1 = c1 * cd1 - s1 * sd1;
                s1 = t1;
            }
        }
        __syncthreads();

        // ---- A fragment: A[m][k] = phi(k, rowbase+m); halves [8h,8h+8) and
        //      [16+8h,..+8) of row rowbase+m  -> two b128 loads --------------
        const v8h* rowA = (const v8h*)&sphiT[rowbase + mn][0];
        v8h a_lo = rowA[half];
        v8h a_hi = rowA[2 + half];
        v16h a = __builtin_shufflevector(a_lo, a_hi,
                 0, 1, 2, 3, 4, 5, 6, 7, 8, 9, 10, 11, 12, 13, 14, 15);

        // ---- 8 col tiles: B[k][nc] = phi(k, colbase+nc); halves [8h,8h+8)
        //      and [16+8h,..+8) of row colbase+nc -> two b128 loads ----------
#pragma unroll
        for (int t = 0; t < 8; t++) {
            int colbase = (ch * 8 + t) * 16;
            const v8h* rowB = (const v8h*)&sphiT[colbase + mn][0];
            v8h b_lo = rowB[half];
            v8h b_hi = rowB[2 + half];
            v16h b = __builtin_shufflevector(b_lo, b_hi,
                     0, 1, 2, 3, 4, 5, 6, 7, 8, 9, 10, 11, 12, 13, 14, 15);
            acc[t] = __builtin_amdgcn_wmma_f32_16x16x32_f16(
                false, a, false, b, (short)0, acc[t], false, false);
        }
        __syncthreads();
    }

    // ---- merge K-split partials into Z -----------------------------------
#pragma unroll
    for (int t = 0; t < 8; t++) {
        int colbase = (ch * 8 + t) * 16;
#pragma unroll
        for (int r = 0; r < 8; r++) {
            int row = rowbase + r + 8 * half;
            atomicAdd(&Zg[row * 256 + colbase + mn], acc[t][r]);
        }
    }
}

// ---------------------------------------------------------------------------
// 4) phiTy[j] = sum_n phi[n][j] * y[n]   (LDS-staged broadcast of x,y)
// ---------------------------------------------------------------------------
__global__ void __launch_bounds__(256)
gp_phity_kernel(const float* __restrict__ m_train, const float* __restrict__ y_train,
                float* ws, int n) {
    __shared__ float sx[256];
    __shared__ float sy[256];
    int t = threadIdx.x;
    float L = ws[WS_L], alpha = ws[WS_ALPHA], isl = ws[WS_ISL];
    float jf  = (float)(t + 1);
    float acc = 0.0f;
    int per  = n / gridDim.x;
    int base = blockIdx.x * per;
    for (int c = 0; c < per; c += 256) {
        sx[t] = m_train[base + c + t];
        sy[t] = y_train[base + c + t];
        __syncthreads();
#pragma unroll 8
        for (int k = 0; k < 256; k++) {
            float theta = alpha * (sx[k] + L);
            acc += __sinf(jf * theta) * sy[k];
        }
        __syncthreads();
    }
    atomicAdd(&ws[PHITY_OFF + t], acc * isl);
}

// ---------------------------------------------------------------------------
// 5) tail: add sn^2*inv_lambda to diag(Z), Cholesky, solve, assemble NLML
// ---------------------------------------------------------------------------
__global__ void __launch_bounds__(256)
gp_solve_kernel(const float* sigma_f, const float* lengthscale, const float* sigma_n,
                float* ws, float* out, int n) {
    __shared__ float s_col[256];
    __shared__ float s_b[256];
    __shared__ float s_red[256];
    __shared__ float s_diag;
    __shared__ float s_logdet;

    int t = threadIdx.x;
    float* Z = ws + Z_OFF;
    const float* invlam = ws + INVLAM_OFF;
    const float* phiTy  = ws + PHITY_OFF;
    float sn2 = sigma_n[0] * sigma_n[0];

    Z[t * 256 + t] += sn2 * invlam[t];
    if (t == 0) s_logdet = 0.0f;
    __syncthreads();

    // in-place Cholesky (lower), right-looking
    for (int k = 0; k < 256; k++) {
        if (t == 0) {
            float d = sqrtf(Z[k * 256 + k]);
            Z[k * 256 + k] = d;
            s_diag = 1.0f / d;
            s_logdet += logf(d);
        }
        __syncthreads();
        int i = k + 1 + t;
        if (i < 256) Z[i * 256 + k] *= s_diag;
        __syncthreads();
        s_col[t] = (t > k) ? Z[t * 256 + k] : 0.0f;
        __syncthreads();
        if (i < 256) {
            float lik = s_col[i];
            for (int j = k + 1; j <= i; j++) Z[i * 256 + j] -= lik * s_col[j];
        }
        __syncthreads();
    }

    // forward solve L w = phiTy
    s_b[t] = phiTy[t];
    __syncthreads();
    for (int k = 0; k < 256; k++) {
        if (t == 0) s_b[k] = s_b[k] / Z[k * 256 + k];
        __syncthreads();
        float wk = s_b[k];
        if (t > k) s_b[t] -= Z[t * 256 + k] * wk;
        __syncthreads();
    }
    // backward solve L^T v = w
    for (int k = 255; k >= 0; k--) {
        if (t == 0) s_b[k] = s_b[k] / Z[k * 256 + k];
        __syncthreads();
        float vk = s_b[k];
        if (t < k) s_b[t] -= Z[k * 256 + t] * vk;
        __syncthreads();
    }

    // vdot = v . phiTy
    s_red[t] = s_b[t] * phiTy[t];
    for (int o = 128; o > 0; o >>= 1) { __syncthreads(); if (t < o) s_red[t] += s_red[t + o]; }
    __syncthreads();
    float vdot = s_red[0];
    __syncthreads();

    // sum log inv_lambda
    s_red[t] = logf(invlam[t]);
    for (int o = 128; o > 0; o >>= 1) { __syncthreads(); if (t < o) s_red[t] += s_red[t + o]; }
    __syncthreads();

    if (t == 0) {
        float sumloginvlam = s_red[0];
        float yTy  = ws[WS_YTY];
        float logQ = (float)(n - GP_M) * logf(sn2) + 2.0f * s_logdet - sumloginvlam;
        float yQiy = (yTy - vdot) / sn2;
        out[0] = 0.5f * logQ + 0.5f * yQiy;
    }
    (void)sigma_f; (void)lengthscale;
}

// ---------------------------------------------------------------------------
extern "C" void kernel_launch(void* const* d_in, const int* in_sizes, int n_in,
                              void* d_out, int out_size, void* d_ws, size_t ws_size,
                              hipStream_t stream) {
    const float* sigma_f     = (const float*)d_in[0];
    const float* lengthscale = (const float*)d_in[1];
    const float* sigma_n     = (const float*)d_in[2];
    const float* m_train     = (const float*)d_in[3];
    const float* y_train     = (const float*)d_in[4];
    float* out = (float*)d_out;
    float* ws  = (float*)d_ws;
    const int n = in_sizes[3];  // 131072

    gp_init_kernel<<<(WS_TOTAL + 255) / 256, 256, 0, stream>>>(ws);
    gp_reduce_kernel<<<512, 256, 0, stream>>>(m_train, y_train, ws, n);
    gp_prep_kernel<<<1, 256, 0, stream>>>(sigma_f, lengthscale, sigma_n, ws);
    dim3 ggrid(32, 4);  // 32-way K split x 4 Z quadrants
    gp_gemm_kernel<<<ggrid, 256, 0, stream>>>(m_train, ws, n);
    gp_phity_kernel<<<64, 256, 0, stream>>>(m_train, y_train, ws, n);
    gp_solve_kernel<<<1, 256, 0, stream>>>(sigma_f, lengthscale, sigma_n, ws, out, n);
    (void)n_in; (void)out_size; (void)ws_size;
}
